// VectorQuantizer_87351044866331
// MI455X (gfx1250) — matrix operand order repaired and checked by reading, not compiled, not run on hardware
//
#include <hip/hip_runtime.h>
#include <hip/hip_bf16.h>

typedef __attribute__((ext_vector_type(16))) __bf16 v16bf;
typedef __attribute__((ext_vector_type(8)))  float  v8f;

#define N_TOK   131072      // B*H*W = 32*64*64
#define K_CODES 512
#define D_DIM   64
#define N_ELEM  8388608     // B*D*H*W

// ---------------------------------------------------------------------------
// Prep 1: per-code squared norms (fp32), zero counts + accumulators. 1x512.
// ---------------------------------------------------------------------------
__global__ void vq_prep_norms_kernel(const float* __restrict__ emb,
                                     float* __restrict__ wsNorm,
                                     int* __restrict__ wsCounts,
                                     float* __restrict__ wsSSE,
                                     float* __restrict__ wsDist) {
    int k = threadIdx.x;            // 0..511
    float s = 0.f;
    const float* e = emb + k * D_DIM;
#pragma unroll
    for (int d = 0; d < D_DIM; ++d) { float v = e[d]; s += v * v; }
    wsNorm[k]   = s;
    wsCounts[k] = 0;
    if (k == 0) { wsSSE[0] = 0.f; wsDist[0] = 0.f; }
}

// ---------------------------------------------------------------------------
// Prep 2: split emb into bf16 hi/lo and repack into WMMA B-fragment order.
//   B 32x16 bf16 fragment (contiguous k order used here): lane L, element e:
//     K = 16*(L>=16) + e,  N = L%16
//   wsB*[((ct*2 + chunk)*32 + lane)*16 + e] = part( emb[n*64 + 32*chunk + K] )
//   128 blocks x 256 = 32768 threads exactly.
// ---------------------------------------------------------------------------
__global__ void vq_repack_kernel(const float* __restrict__ emb,
                                 __bf16* __restrict__ wsBhi,
                                 __bf16* __restrict__ wsBlo) {
    int f = blockIdx.x * 256 + threadIdx.x;      // 0..32767
    int e    = f & 15;
    int lane = (f >> 4) & 31;
    int c    = (f >> 9) & 1;
    int ct   = f >> 10;
    int n = ct * 16 + (lane & 15);
    int d = 32 * c + 16 * (lane >> 4) + e;
    float x = emb[n * D_DIM + d];
    __bf16 xh = (__bf16)x;
    float  xl = x - (float)xh;
    wsBhi[f] = xh;
    wsBlo[f] = (__bf16)xl;
}

// ---------------------------------------------------------------------------
// Main: bf16x3 split-GEMM distances + argmin.  One wave per 16-token tile.
//   512 blocks x 512 threads (16 waves/block) = 8192 waves = 8192 tiles.
//   Dynamic LDS: sBhi 64KB | sBlo 64KB | sNorm 2KB  = 133120 bytes.
// ---------------------------------------------------------------------------
__global__ __launch_bounds__(512) void vq_argmin_kernel(
        const float* __restrict__ z_e,
        const __bf16* __restrict__ wsBhi,
        const __bf16* __restrict__ wsBlo,
        const float* __restrict__ wsNorm,
        int*   __restrict__ wsIdx,
        int*   __restrict__ wsCounts,
        float* __restrict__ wsDist,
        float* __restrict__ idxOutF) {
    extern __shared__ char smem[];
    __bf16* sBhi  = (__bf16*)smem;                   // 32768 bf16
    __bf16* sBlo  = (__bf16*)(smem + 65536);         // 32768 bf16
    float*  sNorm = (float*)(smem + 131072);         // 512 f32

    // ---- stage B fragments + norms into LDS (once per workgroup) ----
    {
        const float4* gh = (const float4*)wsBhi;     // 4096 x 16B = 64KB
        const float4* gl = (const float4*)wsBlo;
        float4* sh = (float4*)sBhi;
        float4* sl = (float4*)sBlo;
        for (int i = threadIdx.x; i < 4096; i += 512) sh[i] = gh[i];
        for (int i = threadIdx.x; i < 4096; i += 512) sl[i] = gl[i];
        if (threadIdx.x < 512) sNorm[threadIdx.x] = wsNorm[threadIdx.x];
    }
    __syncthreads();

    const int lane = threadIdx.x & 31;
    const int wave = threadIdx.x >> 5;
    const int T    = blockIdx.x * 16 + wave;  // tile id, 0..8191
    const int t    = lane & 15;               // token-within-tile / column id
    const int hi   = lane >> 4;

    // tile -> (b, h, w0); tokens are 16 consecutive w at fixed (b,h)
    const int w0 = (T & 3) << 4;
    const int h  = (T >> 2) & 63;
    const int b  = T >> 8;

    // z_e layout [B, D, H, W]: elem(b,d,h,w) at ((b*64+d)*64+h)*64+w
    const float* zb = z_e + ((size_t)b << 18) + ((size_t)h << 6) + (size_t)(w0 + t);

    // A fragments (16x32 bf16, contiguous k order, same as B): lane L, element e holds
    //   K = 16*hi + e
    // chunk c covers dims 32c..32c+31.  Split z into bf16 hi + bf16 residual.
    v16bf ah[2], al[2];
    float part = 0.f;
#pragma unroll
    for (int c = 0; c < 2; ++c) {
#pragma unroll
        for (int e = 0; e < 16; ++e) {
            int k = 16 * hi + e;
            int d = 32 * c + k;
            float x = zb[(size_t)d * 4096];
            __bf16 xh = (__bf16)x;
            float  xl = x - (float)xh;
            ah[c][e] = xh;
            al[c][e] = (__bf16)xl;
            part += x * x;
        }
    }
    // full ||z_t||^2 per token (each half-wave holds half the dims)
    float zn = part + __shfl_xor(part, 16, 32);
    // broadcast to the C-fragment row this lane owns: M = r + 8*hi
    float zn8[8];
#pragma unroll
    for (int r = 0; r < 8; ++r) zn8[r] = __shfl(zn, r + 8 * hi, 32);

    float bestD[8]; int bestI[8];
#pragma unroll
    for (int r = 0; r < 8; ++r) { bestD[r] = 3.4e38f; bestI[r] = 0; }

    for (int ct = 0; ct < 32; ++ct) {
        float en = sNorm[ct * 16 + t];               // ||e_n||^2, n = ct*16 + t
        const int base = ct * 1024 + lane * 16;      // chunk stride = 512 bf16
        v16bf bh0 = *(const v16bf*)(sBhi + base);
        v16bf bh1 = *(const v16bf*)(sBhi + base + 512);
        v16bf bl0 = *(const v16bf*)(sBlo + base);
        v16bf bl1 = *(const v16bf*)(sBlo + base + 512);
        v8f c8 = {};
        // hi*hi
        c8 = __builtin_amdgcn_wmma_f32_16x16x32_bf16(false, ah[0], false, bh0,
                                                     (short)0, c8, false, false);
        c8 = __builtin_amdgcn_wmma_f32_16x16x32_bf16(false, ah[1], false, bh1,
                                                     (short)0, c8, false, false);
        // hi*lo
        c8 = __builtin_amdgcn_wmma_f32_16x16x32_bf16(false, ah[0], false, bl0,
                                                     (short)0, c8, false, false);
        c8 = __builtin_amdgcn_wmma_f32_16x16x32_bf16(false, ah[1], false, bl1,
                                                     (short)0, c8, false, false);
        // lo*hi
        c8 = __builtin_amdgcn_wmma_f32_16x16x32_bf16(false, al[0], false, bh0,
                                                     (short)0, c8, false, false);
        c8 = __builtin_amdgcn_wmma_f32_16x16x32_bf16(false, al[1], false, bh1,
                                                     (short)0, c8, false, false);
        // partial distance (||z||^2 added after the loop; constant per token)
#pragma unroll
        for (int r = 0; r < 8; ++r) {
            float dist = en - 2.0f * c8[r];
            int   idx  = ct * 16 + t;
            bool better = dist < bestD[r];           // strict: lowest index per lane
            bestD[r] = better ? dist : bestD[r];
            bestI[r] = better ? idx  : bestI[r];
        }
    }

    // min across the 16 columns (lanes within each half); tie -> lowest index
#pragma unroll
    for (int r = 0; r < 8; ++r) {
        float d = bestD[r]; int i = bestI[r];
#pragma unroll
        for (int m = 1; m <= 8; m <<= 1) {
            float od = __shfl_xor(d, m, 32);
            int   oi = __shfl_xor(i, m, 32);
            if (od < d || (od == d && oi < i)) { d = od; i = oi; }
        }
        bestD[r] = d + zn8[r];    // full squared distance
        bestI[r] = i;
    }

    if (t == 0) {                       // lanes 0 and 16: tokens 0..7 / 8..15
        const int n0 = T * 16 + 8 * hi;
        float dsum = 0.f;
#pragma unroll
        for (int r = 0; r < 8; ++r) {
            int ki = bestI[r];
            wsIdx[n0 + r]   = ki;
            idxOutF[n0 + r] = (float)ki;
            atomicAdd(&wsCounts[ki], 1);
            dsum += bestD[r];
        }
        atomicAdd(wsDist, dsum);
    }
}

// ---------------------------------------------------------------------------
// Gather z_q (== z_q_st forward value) back to [B,D,H,W]; accumulate exact SSE.
//   32768 blocks x 256 threads = 8388608.
// ---------------------------------------------------------------------------
__global__ void vq_gather_kernel(const float* __restrict__ z_e,
                                 const float* __restrict__ emb,
                                 const int* __restrict__ wsIdx,
                                 float* __restrict__ out,
                                 float* __restrict__ wsSSE) {
    __shared__ float sred[256];
    int e  = blockIdx.x * 256 + threadIdx.x;     // flat [B,D,H,W] index
    int w  = e & 63;
    int hh = (e >> 6) & 63;
    int d  = (e >> 12) & 63;
    int bb = e >> 18;
    int n  = (((bb << 6) + hh) << 6) + w;        // token id in (b,h,w) order
    int k  = wsIdx[n];
    float q = emb[k * D_DIM + d];
    float z = z_e[e];
    out[e] = q;
    float diff = q - z;
    sred[threadIdx.x] = diff * diff;
    __syncthreads();
    for (int s = 128; s > 0; s >>= 1) {
        if (threadIdx.x < s) sred[threadIdx.x] += sred[threadIdx.x + s];
        __syncthreads();
    }
    if (threadIdx.x == 0) atomicAdd(wsSSE, sred[0]);
}

// ---------------------------------------------------------------------------
// Final scalars.  1 block x 512.
// ---------------------------------------------------------------------------
__global__ void vq_stats_kernel(const int* __restrict__ wsCounts,
                                const float* __restrict__ wsSSE,
                                const float* __restrict__ wsDist,
                                float* __restrict__ outs) {
    __shared__ float sent[512];
    __shared__ int   sused[512];
    int k = threadIdx.x;
    int c = wsCounts[k];
    float p = (float)c / (float)N_TOK;           // total is always N_TOK
    sent[k]  = (c > 0) ? (-p * logf(p)) : 0.f;
    sused[k] = (c > 0) ? 1 : 0;
    __syncthreads();
    for (int s = 256; s > 0; s >>= 1) {
        if (k < s) { sent[k] += sent[k + s]; sused[k] += sused[k + s]; }
        __syncthreads();
    }
    if (k == 0) {
        // loss_vq = (1 + BETA) * mean((z_q - z_e)^2) * D = 1.25 * SSE / N_TOK
        outs[0] = 1.25f * wsSSE[0] / (float)N_TOK;
        outs[1] = expf(sent[0]);                      // perplexity
        outs[2] = (float)sused[0];                    // codes_used
        outs[3] = (float)sused[0] / (float)K_CODES;   // usage_ratio
        outs[4] = wsDist[0] / (float)N_TOK;           // avg_dist2
    }
}

// ---------------------------------------------------------------------------
extern "C" void kernel_launch(void* const* d_in, const int* in_sizes, int n_in,
                              void* d_out, int out_size, void* d_ws, size_t ws_size,
                              hipStream_t stream) {
    const float* z_e = (const float*)d_in[0];    // 8388608 f32
    const float* emb = (const float*)d_in[1];    // 32768 f32
    float* out = (float*)d_out;

    // workspace layout (d_ws is 256B aligned)
    __bf16* wsBhi   = (__bf16*)d_ws;                 // 32768 bf16 (64KB)
    __bf16* wsBlo   = wsBhi + 32768;                 // 32768 bf16 (64KB)
    float*  wsNorm  = (float*)(wsBlo + 32768);       // 512 f32
    int*    wsIdx   = (int*)(wsNorm + 512);          // 131072 i32
    int*    wsCounts= wsIdx + N_TOK;                 // 512 i32
    float*  wsSSE   = (float*)(wsCounts + 512);      // 1 f32
    float*  wsDist  = wsSSE + 1;                     // 1 f32

    float* out_zq      = out;                    // 8388608
    float* out_scalars = out + N_ELEM;           // 5 scalars
    float* out_idx     = out + N_ELEM + 5;       // 131072 indices (as f32)

    vq_prep_norms_kernel<<<1, 512, 0, stream>>>(emb, wsNorm, wsCounts, wsSSE, wsDist);
    vq_repack_kernel<<<128, 256, 0, stream>>>(emb, wsBhi, wsBlo);
    vq_argmin_kernel<<<512, 512, 133120, stream>>>(z_e, wsBhi, wsBlo, wsNorm,
                                                   wsIdx, wsCounts, wsDist, out_idx);
    vq_gather_kernel<<<32768, 256, 0, stream>>>(z_e, emb, wsIdx, out_zq, wsSSE);
    vq_stats_kernel<<<1, 512, 0, stream>>>(wsCounts, wsSSE, wsDist, out_scalars);
}
